// NeRFW_89197880803768
// MI455X (gfx1250) — compile-verified, matrix operand order checked by reading, not run on hardware
//
#include <hip/hip_runtime.h>
#include <hip/hip_bf16.h>

// ---------------------------------------------------------------------------
// NeRF-W fused forward for gfx1250 (MI455X): f16 WMMA, f32 accum.
//   pass 1: pack all weights to f16, transposed [doutP][KP] (+f32 bias), once
//   pass 2: one fused kernel, 128-point tiles, all activations in LDS
// Round 2: A-fragments hoisted to VGPRs (loaded once per layer), n-loop
//          processes 2 n-tiles per iteration -> 2 independent WMMA acc chains.
// ---------------------------------------------------------------------------

typedef __attribute__((ext_vector_type(8)))  _Float16 v8h;
typedef __attribute__((ext_vector_type(16))) _Float16 v16h;
typedef __attribute__((ext_vector_type(8)))  float    v8f;

#define XCOLS 161
#define TILE_M 128
#define NTHREADS 256

// ---- logical GEMM matrices (packed layout) --------------------------------
enum { M_LIGHT0=0, M_LIGHT1, M_ENC0, M_ENC1, M_ENC2, M_ENC3, M_ENC4, M_ENC5,
       M_ENC6, M_ENC7, M_DECSIG, M_FINAL, M_DIFF0, M_DIFF1, M_SPEC0, M_SPEC1,
       M_T0, M_T1, M_T2, M_T3, M_THEADS, M_COUNT };

constexpr int KPs[M_COUNT] = {32,32,64,256,256,256,320,256,256,256,
                              256,256,384,128,384,128,384,128,128,128,128};
constexpr int NPs[M_COUNT] = {32,16,256,256,256,256,256,256,256,256,
                              16,256,128,16,128,16,128,128,128,128,16};

constexpr long woffset(int m){ long o=0; for(int i=0;i<m;i++) o += (long)KPs[i]*NPs[i]; return o; }
constexpr long boffset(int m){ long o=0; for(int i=0;i<m;i++) o += NPs[i]; return o; }
constexpr long WTOT = woffset(M_COUNT);   // halfs
constexpr long BTOT = boffset(M_COUNT);   // floats

// ---- source params (order = python dict insertion order) ------------------
enum { S_ENC0=0,S_ENC1,S_ENC2,S_ENC3,S_ENC4,S_ENC5,S_ENC6,S_ENC7,S_FINAL,
       S_DECSIG,S_LIGHT0,S_LIGHT1,S_DIFF0,S_DIFF1,S_SPEC0,S_SPEC1,
       S_T0,S_T1,S_T2,S_T3,S_TSIG,S_TRGB,S_TBETA, S_COUNT };

// ---------------------------------------------------------------------------
// Weight packing: dst[(n+nOff)*KP + rowMap(mode,j)] = W[j*dout+n]  (f32->f16)
// rowMap embeds the concat layouts so concats become plain GEMMs.
// unified decoder feature layout (K=384):
//   [0..47]=a_emb [48..74]=dir [75..79]=0 [80..95]=lf [96..111]=t_emb
//   [112..367]=sse [368..383]=0
// ---------------------------------------------------------------------------
__device__ __forceinline__ int rowMap(int mode, int j) {
  switch (mode) {
    case 1: return (j < 63) ? j : 64 + (j - 63);                       // enc4: [xyz|h]
    case 2: return (j < 48) ? j : ((j < 64) ? 80 + (j - 48)            // diff0: [a|lf|sse]
                                            : 112 + (j - 64));
    case 3: return (j < 75) ? j : ((j < 91) ? 80 + (j - 75)            // spec0: [a|dir|lf|sse]
                                            : 112 + (j - 91));
    case 4: return (j < 16) ? 96 + j : 112 + (j - 16);                 // t0: [t_emb|sse]
    default: return j;
  }
}

__global__ void zero_ws_k(_Float16* w, long nw, float* b, long nb) {
  long i = (long)blockIdx.x * blockDim.x + threadIdx.x;
  long stride = (long)gridDim.x * blockDim.x;
  for (long k = i; k < nw; k += stride) w[k] = (_Float16)0.f;
  for (long k = i; k < nb; k += stride) b[k] = 0.f;
}

__global__ void pack_w_k(const float* __restrict__ W, int din, int dout,
                         _Float16* __restrict__ dst, int KP, int nOff, int mode) {
  long i = (long)blockIdx.x * blockDim.x + threadIdx.x;
  long tot = (long)din * dout;
  if (i >= tot) return;
  int j = (int)(i / dout);
  int n = (int)(i - (long)j * dout);
  dst[(size_t)(n + nOff) * KP + rowMap(mode, j)] = (_Float16)W[i];
}

__global__ void pack_b_k(const float* __restrict__ b, int dout,
                         float* __restrict__ dst, int nOff) {
  int n = blockIdx.x * blockDim.x + threadIdx.x;
  if (n < dout) dst[n + nOff] = b[n];
}

// ---------------------------------------------------------------------------
// WMMA helpers (CDNA5 16x16x32 f16, wave32)
// A layout (16-bit, 16x32): lanes 0-15 row M=lane, K {k0..k0+7, k0+16..k0+23};
// lanes 16-31 same row, K {+8}. B mirrored with N per lane (weights packed
// transposed so both sides are contiguous 16B chunks per lane).
// ---------------------------------------------------------------------------
__device__ __forceinline__ v16h frag16(const _Float16* p) {
  v8h lo = *(const v8h*)p;
  v8h hi = *(const v8h*)(p + 16);
  return __builtin_shufflevector(lo, hi, 0,1,2,3,4,5,6,7,8,9,10,11,12,13,14,15);
}

__device__ __forceinline__ v8f wmma_f16(v16h a, v16h b, v8f c) {
  return __builtin_amdgcn_wmma_f32_16x16x32_f16(false, a, false, b, (short)0, c, false, false);
}

__device__ __forceinline__ float sigmoidf_(float v) { return 1.f / (1.f + __expf(-v)); }
__device__ __forceinline__ float softplusf_(float v) { return (v > 20.f) ? v : log1pf(__expf(v)); }

// generic GEMM over the wave's 16-row stripe: O = act(A @ Wt^T + bias), f16 out to LDS
// A fragments loaded once into VGPRs; 2 n-tiles in flight for WMMA pipelining.
template<int KP, int NP, int ACT>
__device__ __forceinline__ void gemm_lds(const _Float16* __restrict__ A, int lda,
                                         const _Float16* __restrict__ W,
                                         const float* __restrict__ Bias,
                                         _Float16* __restrict__ O, int ldo,
                                         int wave, int lane) {
  constexpr int NK = KP / 32;
  const int r16  = lane & 15;
  const int koff = (lane & 16) ? 8 : 0;
  const _Float16* aRow = A + (size_t)(wave * 16 + r16) * lda + koff;
  const int rb = wave * 16 + ((lane >> 4) & 1) * 8;

  v16h aF[NK];
  #pragma unroll
  for (int k = 0; k < NK; ++k) aF[k] = frag16(aRow + 32 * k);

  if constexpr ((NP & 31) == 0) {
    #pragma unroll 1
    for (int n0 = 0; n0 < NP; n0 += 32) {
      const int c0 = n0 + r16, c1 = n0 + 16 + r16;
      const _Float16* w0 = W + (size_t)c0 * KP + koff;
      const _Float16* w1 = W + (size_t)c1 * KP + koff;
      float b0 = Bias[c0], b1 = Bias[c1];
      v8f acc0, acc1;
      #pragma unroll
      for (int i = 0; i < 8; ++i) { acc0[i] = b0; acc1[i] = b1; }
      #pragma unroll
      for (int k = 0; k < NK; ++k) {
        v16h bb0 = frag16(w0 + 32 * k);
        v16h bb1 = frag16(w1 + 32 * k);
        acc0 = wmma_f16(aF[k], bb0, acc0);
        acc1 = wmma_f16(aF[k], bb1, acc1);
      }
      #pragma unroll
      for (int r = 0; r < 8; ++r) {
        float v0 = acc0[r], v1 = acc1[r];
        if (ACT) { v0 = v0 > 0.f ? v0 : 0.f; v1 = v1 > 0.f ? v1 : 0.f; }
        O[(size_t)(rb + r) * ldo + c0] = (_Float16)v0;
        O[(size_t)(rb + r) * ldo + c1] = (_Float16)v1;
      }
    }
  } else {  // NP == 16 (light1)
    const int col = r16;
    const _Float16* wRow = W + (size_t)col * KP + koff;
    float bv = Bias[col];
    v8f acc;
    #pragma unroll
    for (int i = 0; i < 8; ++i) acc[i] = bv;
    #pragma unroll
    for (int k = 0; k < NK; ++k) acc = wmma_f16(aF[k], frag16(wRow + 32 * k), acc);
    #pragma unroll
    for (int r = 0; r < 8; ++r) {
      float v = acc[r];
      if (ACT) v = v > 0.f ? v : 0.f;
      O[(size_t)(rb + r) * ldo + col] = (_Float16)v;
    }
  }
}

enum { H_DECSIG, H_DIFF, H_SPEC, H_THEADS };

// head GEMM: single 16-wide n-tile, custom epilogue writing f32 to global/LDS
template<int KP, int HEAD>
__device__ __forceinline__ void gemm_head(const _Float16* __restrict__ A, int lda,
                                          const _Float16* __restrict__ W,
                                          const float* __restrict__ Bias,
                                          const float* __restrict__ xg,
                                          float* __restrict__ out,
                                          float* __restrict__ diffBuf,
                                          long m0, long npts, int wave, int lane) {
  constexpr int NK = KP / 32;
  const int r16  = lane & 15;
  const int koff = (lane & 16) ? 8 : 0;
  const _Float16* aRow = A + (size_t)(wave * 16 + r16) * lda + koff;
  const int col = r16;
  const _Float16* wRow = W + (size_t)col * KP + koff;
  float bv = Bias[col];
  v8f acc;
  #pragma unroll
  for (int i = 0; i < 8; ++i) acc[i] = bv;
  #pragma unroll
  for (int k = 0; k < NK; ++k)
    acc = wmma_f16(frag16(aRow + 32 * k), frag16(wRow + 32 * k), acc);
  const int rb = wave * 16 + ((lane >> 4) & 1) * 8;
  #pragma unroll
  for (int r = 0; r < 8; ++r) {
    int row = rb + r;
    long grow = m0 + row;
    if (grow >= npts) continue;
    float v = acc[r];
    if (HEAD == H_DECSIG) {
      if (col == 0) out[grow] = softplusf_(v);                       // static_sigma
    } else if (HEAD == H_DIFF) {
      if (col < 3) diffBuf[row * 4 + col] = sigmoidf_(v);            // diffuse -> LDS
    } else if (HEAD == H_SPEC) {
      if (col < 3) {
        float cl = xg[grow * XCOLS + 160];                           // cloudiness
        out[npts + grow * 3 + col] = diffBuf[row * 4 + col] + (1.f - cl) * sigmoidf_(v);
      }
    } else {                                                         // THEADS
      if (col == 0)      out[4 * npts + grow] = softplusf_(v);       // t_sigma
      else if (col < 4)  out[5 * npts + grow * 3 + (col - 1)] = sigmoidf_(v); // t_rgb
      else if (col == 4) out[8 * npts + grow] = softplusf_(v);       // t_beta
    }
  }
}

// ---------------------------------------------------------------------------
// Fused forward: 128-point tile, 8 waves x 16-row stripes (all wave-local LDS)
// LDS: cat0/cat1 128x320 (xyz cols 0..63, h cols 64..319), dec 128x384,
//      diffBuf 128x4 f32  => ~258 KB (dynamic shared, <320 KB WGP budget)
// ---------------------------------------------------------------------------
__global__ __launch_bounds__(NTHREADS)
void nerfw_fused(const float* __restrict__ x,
                 const _Float16* __restrict__ wW, const float* __restrict__ wB,
                 float* __restrict__ out, int npts_i) {
  extern __shared__ _Float16 smem[];
  _Float16* cat0 = smem;                       // 128*320
  _Float16* cat1 = cat0 + TILE_M * 320;        // 128*320
  _Float16* dec  = cat1 + TILE_M * 320;        // 128*384
  float* diffBuf = (float*)(dec + TILE_M * 384);

  const long npts = npts_i;
  const int tid = threadIdx.x, lane = tid & 31, wave = tid >> 5;
  const long m0 = (long)blockIdx.x * TILE_M;

  // zero-fill pad columns
  for (int r = tid; r < TILE_M; r += NTHREADS) {
    cat0[r * 320 + 63] = (_Float16)0.f;
    cat1[r * 320 + 63] = (_Float16)0.f;
    _Float16* d = dec + r * 384;
    #pragma unroll
    for (int c = 75; c < 80; ++c) d[c] = (_Float16)0.f;
    for (int c = 119; c < 144; ++c) d[c] = (_Float16)0.f;   // light K-pad
    #pragma unroll
    for (int c = 368; c < 384; ++c) d[c] = (_Float16)0.f;
  }
  // coalesced x-tile load, scatter f16 into segment layout
  for (int i = tid; i < TILE_M * XCOLS; i += NTHREADS) {
    int r = i / XCOLS, c = i - r * XCOLS;
    long grow = m0 + r;
    float v = (grow < npts) ? x[grow * XCOLS + c] : 0.f;
    _Float16 h = (_Float16)v;
    if (c < 63)       { cat0[r * 320 + c] = h; cat1[r * 320 + c] = h; }  // xyz
    else if (c < 90)  dec[r * 384 + 48 + (c - 63)]  = h;                 // dir
    else if (c < 138) dec[r * 384 + (c - 90)]       = h;                 // a_emb
    else if (c < 154) dec[r * 384 + 96 + (c - 138)] = h;                 // t_emb
    else              dec[r * 384 + 112 + (c - 154)] = h;                // light raw (temp)
  }
  __syncthreads();

  // light MLP (temp in dec cols 112..175; overwritten later by sse)
  gemm_lds<32,32,1>(dec + 112, 384, wW + woffset(M_LIGHT0), wB + boffset(M_LIGHT0), dec + 144, 384, wave, lane);
  __syncthreads();
  gemm_lds<32,16,0>(dec + 144, 384, wW + woffset(M_LIGHT1), wB + boffset(M_LIGHT1), dec + 80,  384, wave, lane);
  __syncthreads();

  // encoder (ping-pong h in cat0/cat1 cols 64..319; enc4 reads full [xyz|h])
  gemm_lds<64, 256,1>(cat0,      320, wW + woffset(M_ENC0), wB + boffset(M_ENC0), cat1 + 64, 320, wave, lane);
  __syncthreads();
  gemm_lds<256,256,1>(cat1 + 64, 320, wW + woffset(M_ENC1), wB + boffset(M_ENC1), cat0 + 64, 320, wave, lane);
  __syncthreads();
  gemm_lds<256,256,1>(cat0 + 64, 320, wW + woffset(M_ENC2), wB + boffset(M_ENC2), cat1 + 64, 320, wave, lane);
  __syncthreads();
  gemm_lds<256,256,1>(cat1 + 64, 320, wW + woffset(M_ENC3), wB + boffset(M_ENC3), cat0 + 64, 320, wave, lane);
  __syncthreads();
  gemm_lds<320,256,1>(cat0,      320, wW + woffset(M_ENC4), wB + boffset(M_ENC4), cat1 + 64, 320, wave, lane);
  __syncthreads();
  gemm_lds<256,256,1>(cat1 + 64, 320, wW + woffset(M_ENC5), wB + boffset(M_ENC5), cat0 + 64, 320, wave, lane);
  __syncthreads();
  gemm_lds<256,256,1>(cat0 + 64, 320, wW + woffset(M_ENC6), wB + boffset(M_ENC6), cat1 + 64, 320, wave, lane);
  __syncthreads();
  gemm_lds<256,256,1>(cat1 + 64, 320, wW + woffset(M_ENC7), wB + boffset(M_ENC7), cat0 + 64, 320, wave, lane);
  __syncthreads();

  // heads off final h
  gemm_head<256,H_DECSIG>(cat0 + 64, 320, wW + woffset(M_DECSIG), wB + boffset(M_DECSIG),
                          x, out, diffBuf, m0, npts, wave, lane);
  gemm_lds<256,256,0>(cat0 + 64, 320, wW + woffset(M_FINAL), wB + boffset(M_FINAL), dec + 112, 384, wave, lane);
  __syncthreads();

  // color heads off unified decoder features (K=384)
  gemm_lds<384,128,1>(dec, 384, wW + woffset(M_DIFF0), wB + boffset(M_DIFF0), cat1, 320, wave, lane);
  __syncthreads();
  gemm_head<128,H_DIFF>(cat1, 320, wW + woffset(M_DIFF1), wB + boffset(M_DIFF1),
                        x, out, diffBuf, m0, npts, wave, lane);
  __syncthreads();
  gemm_lds<384,128,1>(dec, 384, wW + woffset(M_SPEC0), wB + boffset(M_SPEC0), cat1, 320, wave, lane);
  __syncthreads();
  gemm_head<128,H_SPEC>(cat1, 320, wW + woffset(M_SPEC1), wB + boffset(M_SPEC1),
                        x, out, diffBuf, m0, npts, wave, lane);
  __syncthreads();

  // transient MLP (reuse cat0/cat1 cols 0..127)
  gemm_lds<384,128,1>(dec,  384, wW + woffset(M_T0), wB + boffset(M_T0), cat0, 320, wave, lane);
  __syncthreads();
  gemm_lds<128,128,1>(cat0, 320, wW + woffset(M_T1), wB + boffset(M_T1), cat1, 320, wave, lane);
  __syncthreads();
  gemm_lds<128,128,1>(cat1, 320, wW + woffset(M_T2), wB + boffset(M_T2), cat0, 320, wave, lane);
  __syncthreads();
  gemm_lds<128,128,1>(cat0, 320, wW + woffset(M_T3), wB + boffset(M_T3), cat1, 320, wave, lane);
  __syncthreads();
  gemm_head<128,H_THEADS>(cat1, 320, wW + woffset(M_THEADS), wB + boffset(M_THEADS),
                          x, out, diffBuf, m0, npts, wave, lane);
}

// ---------------------------------------------------------------------------
// Host
// ---------------------------------------------------------------------------
extern "C" void kernel_launch(void* const* d_in, const int* in_sizes, int n_in,
                              void* d_out, int out_size, void* d_ws, size_t ws_size,
                              hipStream_t stream) {
  const float* x = (const float*)d_in[0];
  const int npts = in_sizes[0] / XCOLS;

  // Detect params flatten order: insertion (enc0.W=16128 first) vs sorted
  // keys (dec_sigma.W=256 first, jax tree order).
  const bool sorted = (n_in > 1 && in_sizes[1] == 256);
  static const int sortedPos[S_COUNT] = {3,4,5,6,7,8,9,10,11,0,12,13,1,2,14,15,16,17,18,19,22,21,20};
  const float* Wsrc[S_COUNT];
  const float* Bsrc[S_COUNT];
  for (int s = 0; s < S_COUNT; ++s) {
    int pos = sorted ? sortedPos[s] : s;
    Wsrc[s] = (const float*)d_in[1 + 2 * pos];
    Bsrc[s] = (const float*)d_in[2 + 2 * pos];
  }

  _Float16* wsW = (_Float16*)d_ws;
  float* wsB = (float*)((char*)d_ws + sizeof(_Float16) * WTOT);

  // zero the packed-weight region (pads + concat gaps must be 0)
  zero_ws_k<<<1024, 256, 0, stream>>>(wsW, WTOT, wsB, BTOT);

  struct PD { int src, mat, din, dout, mode, nOff; };
  static const PD pds[] = {
    {S_LIGHT0,M_LIGHT0,  7, 32,0,0}, {S_LIGHT1,M_LIGHT1, 32, 16,0,0},
    {S_ENC0,  M_ENC0,   63,256,0,0}, {S_ENC1,  M_ENC1,  256,256,0,0},
    {S_ENC2,  M_ENC2,  256,256,0,0}, {S_ENC3,  M_ENC3,  256,256,0,0},
    {S_ENC4,  M_ENC4,  319,256,1,0}, {S_ENC5,  M_ENC5,  256,256,0,0},
    {S_ENC6,  M_ENC6,  256,256,0,0}, {S_ENC7,  M_ENC7,  256,256,0,0},
    {S_DECSIG,M_DECSIG,256,  1,0,0}, {S_FINAL, M_FINAL, 256,256,0,0},
    {S_DIFF0, M_DIFF0, 320,128,2,0}, {S_DIFF1, M_DIFF1, 128,  3,0,0},
    {S_SPEC0, M_SPEC0, 347,128,3,0}, {S_SPEC1, M_SPEC1, 128,  3,0,0},
    {S_T0,    M_T0,    272,128,4,0}, {S_T1,    M_T1,    128,128,0,0},
    {S_T2,    M_T2,    128,128,0,0}, {S_T3,    M_T3,    128,128,0,0},
    {S_TSIG,  M_THEADS,128,  1,0,0}, {S_TRGB,  M_THEADS,128,  3,0,1},
    {S_TBETA, M_THEADS,128,  1,0,4},
  };
  for (const PD& p : pds) {
    long tot = (long)p.din * p.dout;
    int blocks = (int)((tot + 255) / 256);
    pack_w_k<<<blocks, 256, 0, stream>>>(Wsrc[p.src], p.din, p.dout,
                                         wsW + woffset(p.mat), KPs[p.mat], p.nOff, p.mode);
    pack_b_k<<<(p.dout + 255) / 256, 256, 0, stream>>>(Bsrc[p.src], p.dout,
                                                       wsB + boffset(p.mat), p.nOff);
  }

  const size_t smemBytes = (size_t)(2 * TILE_M * 320 + TILE_M * 384) * sizeof(_Float16)
                         + (size_t)TILE_M * 4 * sizeof(float);   // ~258 KB
  (void)hipFuncSetAttribute((const void*)nerfw_fused,
                            hipFuncAttributeMaxDynamicSharedMemorySize, (int)smemBytes);

  const int nblk = (npts + TILE_M - 1) / TILE_M;
  nerfw_fused<<<nblk, NTHREADS, smemBytes, stream>>>(x, wsW, wsB, (float*)d_out, npts);
  (void)out_size; (void)ws_size;
}
